// GATClassifier_23330262352277
// MI455X (gfx1250) — matrix operand order repaired and checked by reading, not compiled, not run on hardware
//
#include <hip/hip_runtime.h>
#include <hip/hip_bf16.h>
#include <stdint.h>

// ---------------------------------------------------------------------------
// Types for WMMA
// ---------------------------------------------------------------------------
typedef __attribute__((ext_vector_type(16))) __bf16 v16bf;
typedef __attribute__((ext_vector_type(8)))  float  v8f;
typedef __attribute__((ext_vector_type(4)))  float  f32x4;

// ---------------------------------------------------------------------------
// Device helpers
// ---------------------------------------------------------------------------
__device__ __forceinline__ void atomAddF32(float* p, float v) {
#if defined(__gfx1250__)
  // Hardware FP32 atomic add (no-return form) -> GLOBAL_ATOMIC_ADD_F32
  asm volatile("global_atomic_add_f32 %0, %1, off" :: "v"(p), "v"(v) : "memory");
#else
  atomicAdd(p, v);
#endif
}

// Order-preserving encoding of float into uint so we can use the native
// integer atomicMax (GLOBAL_ATOMIC_MAX_U32) for the segment-max.
__device__ __forceinline__ unsigned ordEnc(float f) {
  unsigned u = __float_as_uint(f);
  return (u & 0x80000000u) ? ~u : (u | 0x80000000u);
}
__device__ __forceinline__ float ordDec(unsigned u) {
  return __uint_as_float((u & 0x80000000u) ? (u ^ 0x80000000u) : ~u);
}

__device__ __forceinline__ void edgeSrcDst(const int* ei, int E, int idx, int& s, int& d) {
  if (idx < E) { s = ei[idx]; d = ei[E + idx]; }
  else         { s = idx - E; d = idx - E; }      // self loops appended
}

// ---------------------------------------------------------------------------
// Weight transpose + fp32 -> bf16 convert: W[K,Nt] row-major -> Wt[Nt][K]
// ---------------------------------------------------------------------------
__global__ void convert_w_kernel(const float* __restrict__ W, __bf16* __restrict__ Wt,
                                 int K, int Nt) {
  int t = blockIdx.x * blockDim.x + threadIdx.x;
  if (t >= K * Nt) return;
  int n = t / K;
  int k = t - n * K;
  Wt[t] = (__bf16)W[(size_t)k * Nt + n];
}

// ---------------------------------------------------------------------------
// bf16 WMMA GEMM: C[M,Nt] = A[M,K](fp32, converted on the fly) * Wt (bf16,
// column-major [Nt][K]).  One wave computes a 16x64 tile with 4 accumulators,
// K-loop in steps of 32 using v_wmma_f32_16x16x32_bf16.
//   A layout (16x32 bf16): lane L holds M = L&15;  16 consecutive K values
//   kb..kb+15 (elem 0..15) with kb = (L>>4)*16.
//   B layout (32x16 bf16): lane L holds N = L&15, 16 consecutive K values
//   starting at (L>>4)*16  -> one 32-byte contiguous load from Wt.
// ---------------------------------------------------------------------------
template<bool RELU_A>
__global__ void gemm_bf16_wmma(const float* __restrict__ A, const __bf16* __restrict__ Wt,
                               float* __restrict__ C, int M, int K, int Nt) {
  const int lane    = threadIdx.x & 31;
  const int wave    = threadIdx.x >> 5;
  const int rowTile = blockIdx.x * 8 + wave;          // wave-uniform guard
  if (rowTile * 16 >= M) return;
  const int colBase = blockIdx.y * 64;
  const int row     = rowTile * 16 + (lane & 15);
  const int kbA     = (lane >> 4) * 16;
  const int kbB     = (lane >> 4) * 16;
  const int nloc    = lane & 15;

  v8f acc[4] = {v8f{}, v8f{}, v8f{}, v8f{}};
  const float* arow = A + (size_t)row * K;

  for (int ks = 0; ks < K; ks += 32) {
    // ---- A fragment: one 64-byte fp32 run, convert to bf16 -------------
    f32x4 a0 = *(const f32x4*)(arow + ks + kbA + 0);
    f32x4 a1 = *(const f32x4*)(arow + ks + kbA + 4);
    f32x4 a2 = *(const f32x4*)(arow + ks + kbA + 8);
    f32x4 a3 = *(const f32x4*)(arow + ks + kbA + 12);
    v16bf afrag;
#pragma unroll
    for (int e = 0; e < 4; ++e) {
      float x0 = a0[e], x1 = a1[e], x2 = a2[e], x3 = a3[e];
      if (RELU_A) { x0 = fmaxf(x0, 0.f); x1 = fmaxf(x1, 0.f);
                    x2 = fmaxf(x2, 0.f); x3 = fmaxf(x3, 0.f); }
      afrag[e]      = (__bf16)x0;
      afrag[4 + e]  = (__bf16)x1;
      afrag[8 + e]  = (__bf16)x2;
      afrag[12 + e] = (__bf16)x3;
    }
    // ---- 4 x (B fragment + WMMA) ---------------------------------------
#pragma unroll
    for (int j = 0; j < 4; ++j) {
      const __bf16* wp = Wt + (size_t)(colBase + j * 16 + nloc) * K + ks + kbB;
      v16bf bfrag = *(const v16bf*)wp;               // 32B contiguous
      acc[j] = __builtin_amdgcn_wmma_f32_16x16x32_bf16(
          false, afrag, false, bfrag, (short)0, acc[j], false, false);
    }
  }

  // ---- store: VGPR r -> M = r + (lane>>4)*8, N = lane&15 -----------------
#pragma unroll
  for (int r = 0; r < 8; ++r) {
    int rr = rowTile * 16 + r + ((lane >> 4) << 3);
#pragma unroll
    for (int j = 0; j < 4; ++j)
      C[(size_t)rr * Nt + colBase + j * 16 + nloc] = acc[j][r];
  }
}

// ---------------------------------------------------------------------------
// Attention projections: a_s[n,h] = sum_c h[n,h,c]*att_s[h,c] (same for a_d).
// One wave per (node, head); shuffle reduction.
// ---------------------------------------------------------------------------
template<int H, int C>
__global__ void attn_scores(const float* __restrict__ hfeat, const float* __restrict__ atts,
                            const float* __restrict__ attd, float* __restrict__ aS,
                            float* __restrict__ aD, int n) {
  int w    = (blockIdx.x * blockDim.x + threadIdx.x) >> 5;
  int lane = threadIdx.x & 31;
  if (w >= n * H) return;
  int node = w / H, hd = w - node * H;
  const float* hp = hfeat + (size_t)node * (H * C) + hd * C;
  float s = 0.f, d = 0.f;
#pragma unroll
  for (int j = 0; j < C / 32; ++j) {
    float hv = hp[j * 32 + lane];
    s += hv * atts[hd * C + j * 32 + lane];
    d += hv * attd[hd * C + j * 32 + lane];
  }
  for (int off = 16; off; off >>= 1) {
    s += __shfl_down(s, off, 32);
    d += __shfl_down(d, off, 32);
  }
  if (lane == 0) { aS[w] = s; aD[w] = d; }
}

// ---------------------------------------------------------------------------
// Per-layer init: m = encoded(-inf)=0, denom = 0, out accumulator = bias.
// ---------------------------------------------------------------------------
__global__ void init_layer(unsigned* __restrict__ m_u, float* __restrict__ denom,
                           float* __restrict__ out, const float* __restrict__ bias,
                           int n, int HC, int H) {
  int t = blockIdx.x * blockDim.x + threadIdx.x;
  if (t < n * HC) out[t] = bias[t & (HC - 1)];       // HC is a power of two
  if (t < n * H) { m_u[t] = 0u; denom[t] = 0.f; }
}

// ---------------------------------------------------------------------------
// Edge pass 1: segment max of LeakyReLU scores via integer atomicMax.
// ---------------------------------------------------------------------------
template<int H>
__global__ void edge_max(const int* __restrict__ ei, int E, int Etot,
                         const float* __restrict__ aS, const float* __restrict__ aD,
                         unsigned* __restrict__ m_u) {
  int idx = blockIdx.x * blockDim.x + threadIdx.x;
  if (idx >= Etot) return;
  int s, d; edgeSrcDst(ei, E, idx, s, d);
#pragma unroll
  for (int h = 0; h < H; ++h) {
    float e = aS[s * H + h] + aD[d * H + h];
    e = e > 0.f ? e : 0.2f * e;
    atomicMax(&m_u[d * H + h], ordEnc(e));
  }
}

// ---------------------------------------------------------------------------
// Edge pass 2: denom[dst] += exp(e - m[dst])
// ---------------------------------------------------------------------------
template<int H>
__global__ void edge_sum(const int* __restrict__ ei, int E, int Etot,
                         const float* __restrict__ aS, const float* __restrict__ aD,
                         const unsigned* __restrict__ m_u, float* __restrict__ denom) {
  int idx = blockIdx.x * blockDim.x + threadIdx.x;
  if (idx >= Etot) return;
  int s, d; edgeSrcDst(ei, E, idx, s, d);
#pragma unroll
  for (int h = 0; h < H; ++h) {
    float e = aS[s * H + h] + aD[d * H + h];
    e = e > 0.f ? e : 0.2f * e;
    float p = __expf(e - ordDec(m_u[d * H + h]));
    atomAddF32(&denom[d * H + h], p);
  }
}

// ---------------------------------------------------------------------------
// Edge pass 3: out[dst] += alpha * h[src].  One wave per edge; alpha computed
// once per head (lane h) and broadcast with shuffles; coalesced 128B chunks.
// ---------------------------------------------------------------------------
template<int H, int C>
__global__ void edge_agg(const int* __restrict__ ei, int E, int Etot,
                         const float* __restrict__ aS, const float* __restrict__ aD,
                         const unsigned* __restrict__ m_u, const float* __restrict__ denom,
                         const float* __restrict__ hfeat, float* __restrict__ out) {
  int w    = (blockIdx.x * blockDim.x + threadIdx.x) >> 5;
  int lane = threadIdx.x & 31;
  if (w >= Etot) return;
  int s, d; edgeSrcDst(ei, E, w, s, d);
  int h = lane & (H - 1);
  float e = aS[s * H + h] + aD[d * H + h];
  e = e > 0.f ? e : 0.2f * e;
  float alpha = __expf(e - ordDec(m_u[d * H + h])) / (denom[d * H + h] + 1e-16f);
  const float* hs = hfeat + (size_t)s * (H * C);
  float*       od = out   + (size_t)d * (H * C);
#pragma unroll
  for (int j = 0; j < (H * C) / 32; ++j) {
    int c = j * 32 + lane;
    float a = __shfl(alpha, c / C, 32);              // head = c / C (compile-time shift)
    atomAddF32(&od[c], hs[c] * a);
  }
}

// ---------------------------------------------------------------------------
// Pool: pooled[b] += relu(out2[n]); cnt[b] += 1.  One wave per node.
// ---------------------------------------------------------------------------
__global__ void zero_pool(float* __restrict__ pooled, float* __restrict__ cnt) {
  int t = blockIdx.x * blockDim.x + threadIdx.x;
  if (t < 16 * 64) pooled[t] = 0.f;
  if (t < 16) cnt[t] = 0.f;
}

__global__ void pool_kernel(const float* __restrict__ out2, const int* __restrict__ batch,
                            float* __restrict__ pooled, float* __restrict__ cnt, int n) {
  int w    = (blockIdx.x * blockDim.x + threadIdx.x) >> 5;
  int lane = threadIdx.x & 31;
  if (w >= n) return;
  int b = batch[w];
#pragma unroll
  for (int j = 0; j < 2; ++j) {
    int c = j * 32 + lane;
    float v = fmaxf(out2[(size_t)w * 64 + c], 0.f);
    atomAddF32(&pooled[b * 64 + c], v);
  }
  if (lane == 0) atomAddF32(&cnt[b], 1.0f);
}

// ---------------------------------------------------------------------------
// Final linear: out[16,10] = (pooled/cnt) @ Wl[64,10] + bl
// ---------------------------------------------------------------------------
__global__ void final_linear(const float* __restrict__ pooled, const float* __restrict__ cnt,
                             const float* __restrict__ Wl, const float* __restrict__ bl,
                             float* __restrict__ out) {
  int t = threadIdx.x;
  if (t >= 16 * 10) return;
  int b = t / 10, o = t - b * 10;
  float inv = 1.0f / fmaxf(cnt[b], 1.0f);
  float acc = bl[o];
#pragma unroll
  for (int c = 0; c < 64; ++c)
    acc += pooled[b * 64 + c] * inv * Wl[c * 10 + o];
  out[t] = acc;
}

// ---------------------------------------------------------------------------
// Host-side launcher
// ---------------------------------------------------------------------------
extern "C" void kernel_launch(void* const* d_in, const int* in_sizes, int n_in,
                              void* d_out, int out_size, void* d_ws, size_t ws_size,
                              hipStream_t stream) {
  const float* x   = (const float*)d_in[0];
  const int*   ei  = (const int*)d_in[1];
  const int*   bat = (const int*)d_in[2];
  const float* W1  = (const float*)d_in[3];
  const float* as1 = (const float*)d_in[4];
  const float* ad1 = (const float*)d_in[5];
  const float* b1  = (const float*)d_in[6];
  const float* W2  = (const float*)d_in[7];
  const float* as2 = (const float*)d_in[8];
  const float* ad2 = (const float*)d_in[9];
  const float* b2  = (const float*)d_in[10];
  const float* Wl  = (const float*)d_in[11];
  const float* bl  = (const float*)d_in[12];
  float* out = (float*)d_out;

  const int N    = in_sizes[0] / 128;   // 50000
  const int E    = in_sizes[1] / 2;     // 800000
  const int Etot = E + N;               // self-loops appended

  // ---- carve workspace ---------------------------------------------------
  char* p = (char*)d_ws;
  auto carve = [&](size_t bytes) -> char* {
    char* r = p;
    p += (bytes + 255) & ~(size_t)255;
    return r;
  };
  __bf16*   W1t    = (__bf16*)  carve((size_t)256 * 128 * 2);
  __bf16*   W2t    = (__bf16*)  carve((size_t)64 * 256 * 2);
  float*    h1     = (float*)   carve((size_t)N * 256 * 4);
  float*    out1   = (float*)   carve((size_t)N * 256 * 4);
  float*    h2     = (float*)   carve((size_t)N * 64 * 4);
  float*    out2   = (float*)   carve((size_t)N * 64 * 4);
  float*    aS1    = (float*)   carve((size_t)N * 4 * 4);
  float*    aD1    = (float*)   carve((size_t)N * 4 * 4);
  unsigned* m1     = (unsigned*)carve((size_t)N * 4 * 4);
  float*    dn1    = (float*)   carve((size_t)N * 4 * 4);
  float*    aS2    = (float*)   carve((size_t)N * 4);
  float*    aD2    = (float*)   carve((size_t)N * 4);
  unsigned* m2     = (unsigned*)carve((size_t)N * 4);
  float*    dn2    = (float*)   carve((size_t)N * 4);
  float*    pooled = (float*)   carve((size_t)16 * 64 * 4);
  float*    cnt    = (float*)   carve(64);
  (void)ws_size; (void)n_in; (void)out_size;

  const int TB = 256;

  // ---- weight conversion -------------------------------------------------
  convert_w_kernel<<<(128 * 256 + TB - 1) / TB, TB, 0, stream>>>(W1, W1t, 128, 256);
  convert_w_kernel<<<(256 * 64 + TB - 1) / TB, TB, 0, stream>>>(W2, W2t, 256, 64);

  // ---- layer 1 -----------------------------------------------------------
  dim3 g1((N / 16 + 7) / 8, 256 / 64);
  gemm_bf16_wmma<false><<<g1, TB, 0, stream>>>(x, W1t, h1, N, 128, 256);
  attn_scores<4, 64><<<((size_t)N * 4 * 32 + TB - 1) / TB, TB, 0, stream>>>(h1, as1, ad1, aS1, aD1, N);
  init_layer<<<((size_t)N * 256 + TB - 1) / TB, TB, 0, stream>>>(m1, dn1, out1, b1, N, 256, 4);
  edge_max<4><<<(Etot + TB - 1) / TB, TB, 0, stream>>>(ei, E, Etot, aS1, aD1, m1);
  edge_sum<4><<<(Etot + TB - 1) / TB, TB, 0, stream>>>(ei, E, Etot, aS1, aD1, m1, dn1);
  edge_agg<4, 64><<<((size_t)Etot * 32 + TB - 1) / TB, TB, 0, stream>>>(
      ei, E, Etot, aS1, aD1, m1, dn1, h1, out1);

  // ---- layer 2 (ReLU of out1 fused into GEMM A-conversion) --------------
  dim3 g2((N / 16 + 7) / 8, 1);
  gemm_bf16_wmma<true><<<g2, TB, 0, stream>>>(out1, W2t, h2, N, 256, 64);
  attn_scores<1, 64><<<((size_t)N * 32 + TB - 1) / TB, TB, 0, stream>>>(h2, as2, ad2, aS2, aD2, N);
  init_layer<<<((size_t)N * 64 + TB - 1) / TB, TB, 0, stream>>>(m2, dn2, out2, b2, N, 64, 1);
  edge_max<1><<<(Etot + TB - 1) / TB, TB, 0, stream>>>(ei, E, Etot, aS2, aD2, m2);
  edge_sum<1><<<(Etot + TB - 1) / TB, TB, 0, stream>>>(ei, E, Etot, aS2, aD2, m2, dn2);
  edge_agg<1, 64><<<((size_t)Etot * 32 + TB - 1) / TB, TB, 0, stream>>>(
      ei, E, Etot, aS2, aD2, m2, dn2, h2, out2);

  // ---- pooling + classifier ---------------------------------------------
  zero_pool<<<4, TB, 0, stream>>>(pooled, cnt);
  pool_kernel<<<((size_t)N * 32 + TB - 1) / TB, TB, 0, stream>>>(out2, bat, pooled, cnt, N);
  final_linear<<<1, TB, 0, stream>>>(pooled, cnt, Wl, bl, out);
}
